// RiemGrassAtt_90220083020128
// MI455X (gfx1250) — matrix operand order repaired and checked by reading, not compiled, not run on hardware
//
#include <hip/hip_runtime.h>
#include <hip/hip_fp16.h>

typedef __attribute__((ext_vector_type(16))) _Float16 v16h;
typedef __attribute__((ext_vector_type(8)))  float    v8f;
typedef __attribute__((ext_vector_type(4)))  unsigned int u32x4;
typedef __attribute__((ext_vector_type(8)))  int          i32x8;
typedef __attribute__((ext_vector_type(4)))  int          i32x4;

#define NB    32
#define SEQ   256
#define DIMC  768
#define NH    12
#define HD    64
#define SLABS (NB * NH)   // 384

// ---------------- WMMA fragment loaders (v_wmma_f32_16x16x32_f16; A/B in this file's contiguous k order) ----
// A (16x32 f16): lane L: m=L%16, half=L/16; elem j -> K = 16*half+j.
// B (32x16 f16): lane L: n=L%16, half=L/16; elem j -> K = 16*half+j.
// C/D (16x16 f32): lane L: n=L%16; VGPR v -> row m = v + 8*(L/16).

__device__ __forceinline__ v16h ldA_f32(const float* __restrict__ base, int ld,
                                        int m0, int k0, int lane) {
  const int m = lane & 15, half = lane >> 4;
  const float* rp = base + (size_t)(m0 + m) * ld + k0;
  v16h a;
#pragma unroll
  for (int j = 0; j < 8; ++j) {
    a[j]     = (_Float16)rp[16 * half + j];
    a[8 + j] = (_Float16)rp[16 * half + 8 + j];
  }
  return a;
}

__device__ __forceinline__ v16h ldA_f16(const _Float16* __restrict__ base, int ld,
                                        int m0, int k0, int lane) {
  const int m = lane & 15, half = lane >> 4;
  const _Float16* rp = base + (size_t)(m0 + m) * ld + k0;
  v16h a;
#pragma unroll
  for (int j = 0; j < 8; ++j) {
    a[j]     = rp[16 * half + j];
    a[8 + j] = rp[16 * half + 8 + j];
  }
  return a;
}

// B[k][n] = W[n][k]  (row n of W holds contiguous k) : computing X * W^T
__device__ __forceinline__ v16h ldBT_f32(const float* __restrict__ w, int ld,
                                         int n0, int k0, int lane) {
  const int n = lane & 15, half = lane >> 4;
  const float* rp = w + (size_t)(n0 + n) * ld + k0 + 16 * half;
  v16h b;
#pragma unroll
  for (int j = 0; j < 16; ++j) b[j] = (_Float16)rp[j];
  return b;
}

__device__ __forceinline__ v16h ldBT_f16(const _Float16* __restrict__ w, int ld,
                                         int n0, int k0, int lane) {
  const int n = lane & 15, half = lane >> 4;
  const _Float16* rp = w + (size_t)(n0 + n) * ld + k0 + 16 * half;
  v16h b;
#pragma unroll
  for (int j = 0; j < 16; ++j) b[j] = rp[j];
  return b;
}

// ---------------- K1: qkv = x @ qkv_w^T + b, scattered to q_f32,k_f32,v_f16 --
__global__ void __launch_bounds__(256)
k_qkv(const float* __restrict__ x, const float* __restrict__ qkv_w,
      const float* __restrict__ qkv_b,
      float* __restrict__ qf, float* __restrict__ kf, _Float16* __restrict__ vh) {
  const int lane = threadIdx.x & 31, wave = threadIdx.x >> 5;
  const int m0 = blockIdx.y * 32 + (wave & 1) * 16;
  const int n0 = blockIdx.x * 64 + (wave >> 1) * 16;
  v8f acc = {};
#pragma unroll 4
  for (int k0 = 0; k0 < DIMC; k0 += 32) {
    if (k0 + 128 < DIMC) {   // pull next K-tiles toward the WGP (global_prefetch_b8)
      __builtin_prefetch(x + (size_t)(m0 + (lane & 15)) * DIMC + k0 + 128, 0, 1);
      __builtin_prefetch(qkv_w + (size_t)(n0 + (lane & 15)) * DIMC + k0 + 128, 0, 1);
    }
    v16h a = ldA_f32(x, DIMC, m0, k0, lane);
    v16h b = ldBT_f32(qkv_w, DIMC, n0, k0, lane);
    acc = __builtin_amdgcn_wmma_f32_16x16x32_f16(false, a, false, b, (short)0, acc, false, false);
  }
  const int half = lane >> 4;
  const int col = n0 + (lane & 15);
  const int which = col / DIMC, rem = col % DIMC;
  const int h = rem / HD, dd = rem % HD;
  const float bias = qkv_b[col];
#pragma unroll
  for (int v = 0; v < 8; ++v) {
    const int r = m0 + v + 8 * half;
    const int b_ = r >> 8, nn = r & 255;
    const size_t idx = (((size_t)(b_ * NH + h) * SEQ) + nn) * HD + dd;
    const float val = acc[v] + bias;
    if (which == 0)      qf[idx] = val;
    else if (which == 1) kf[idx] = val;
    else                 vh[idx] = (_Float16)val;
  }
}

// ---------------- K2: CholeskyQR per 256x64 slab (Gram via WMMA in LDS) -----
__global__ void __launch_bounds__(256)
k_cholqr(const float* __restrict__ qf, const float* __restrict__ kf,
         _Float16* __restrict__ qg, _Float16* __restrict__ kg) {
  __shared__ _Float16 As[SEQ * HD];   // 32 KB: slab in f16
  __shared__ float    G[HD * 65];     // Gram / R factor, padded stride
  const int t = threadIdx.x;
  const int slab = blockIdx.x;
  const float* src = (slab < SLABS) ? qf + (size_t)slab * SEQ * HD
                                    : kf + (size_t)(slab - SLABS) * SEQ * HD;
  _Float16* dst = (slab < SLABS) ? qg + (size_t)slab * SEQ * HD
                                 : kg + (size_t)(slab - SLABS) * SEQ * HD;
  // stage slab row t into LDS
#pragma unroll 8
  for (int i = 0; i < HD; ++i) As[t * HD + i] = (_Float16)src[(size_t)t * HD + i];
  __syncthreads();

  // Gram G = A^T A  (M=N=64, K=256): 16 tiles, 2 per wave
  const int lane = t & 31, wave = t >> 5;
  const int m = lane & 15, half = lane >> 4;
#pragma unroll
  for (int r = 0; r < 2; ++r) {
    const int tile = wave * 2 + r;
    const int i0 = (tile >> 2) * 16, j0 = (tile & 3) * 16;
    v8f acc = {};
    for (int k0 = 0; k0 < SEQ; k0 += 32) {
      v16h a, b;
#pragma unroll
      for (int j = 0; j < 8; ++j) {
        a[j]     = As[(k0 + 16 * half + j) * HD + i0 + m];       // A^T row = A col
        a[8 + j] = As[(k0 + 16 * half + 8 + j) * HD + i0 + m];
      }
#pragma unroll
      for (int j = 0; j < 16; ++j)
        b[j] = As[(k0 + 16 * half + j) * HD + j0 + m];
      acc = __builtin_amdgcn_wmma_f32_16x16x32_f16(false, a, false, b, (short)0, acc, false, false);
    }
#pragma unroll
    for (int v = 0; v < 8; ++v)
      G[(i0 + v + 8 * half) * 65 + j0 + m] = acc[v];
  }
  __syncthreads();

  // Cholesky: G = R^T R (upper R kept in G)
  for (int j = 0; j < HD; ++j) {
    if (t == 0) G[j * 65 + j] = sqrtf(fmaxf(G[j * 65 + j], 1e-20f));
    __syncthreads();
    const float inv = 1.0f / G[j * 65 + j];
    if (t > j && t < HD) G[j * 65 + t] *= inv;
    __syncthreads();
    const int ci = t >> 6, ck = t & 63;
    if (ck > j) {
      const float gjk = G[j * 65 + ck];
      for (int i = j + 1 + ci; i < HD; i += 4)
        G[i * 65 + ck] -= G[j * 65 + i] * gjk;
    }
    __syncthreads();
  }

  // Q = A * R^{-1}: per-row forward substitution, in place in LDS
  for (int i = 0; i < HD; ++i) {
    const float xi = (float)As[t * HD + i] / G[i * 65 + i];
    dst[(size_t)t * HD + i] = (_Float16)xi;
    for (int k = i + 1; k < HD; ++k)
      As[t * HD + k] = (_Float16)((float)As[t * HD + k] - xi * G[i * 65 + k]);
  }
}

// ---------------- K3: attn_g = (Qg Kg^T)^2 * scale  (per slab) --------------
__global__ void __launch_bounds__(256)
k_dots(const _Float16* __restrict__ qg, const _Float16* __restrict__ kg,
       const float* __restrict__ gsc, _Float16* __restrict__ ag) {
  const int lane = threadIdx.x & 31, wave = threadIdx.x >> 5;
  const int slab = blockIdx.z;
  const int m0 = blockIdx.y * 32 + (wave & 1) * 16;
  const int n0 = blockIdx.x * 64 + (wave >> 1) * 16;
  const _Float16* A  = qg + (size_t)slab * SEQ * HD;
  const _Float16* Bm = kg + (size_t)slab * SEQ * HD;
  v8f acc = {};
#pragma unroll
  for (int k0 = 0; k0 < HD; k0 += 32) {
    v16h a = ldA_f16(A, HD, m0, k0, lane);
    v16h b = ldBT_f16(Bm, HD, n0, k0, lane);   // dots[n][m] = sum_d q[n,d]k[m,d]
    acc = __builtin_amdgcn_wmma_f32_16x16x32_f16(false, a, false, b, (short)0, acc, false, false);
  }
  const float s = gsc[0];
  const int half = lane >> 4, col = n0 + (lane & 15);
  _Float16* out = ag + (size_t)slab * SEQ * SEQ;
#pragma unroll
  for (int v = 0; v < 8; ++v) {
    const float d = acc[v];
    out[(size_t)(m0 + v + 8 * half) * SEQ + col] = (_Float16)(d * d * s);
  }
}

// ---------------- K4: 2H->H channel mix + row softmax -----------------------
__global__ void __launch_bounds__(256)
k_mixsoftmax(const _Float16* __restrict__ ag, const float* __restrict__ conv_w,
             const float* __restrict__ conv_b, _Float16* __restrict__ at) {
  __shared__ float red[8];
  const int bo = blockIdx.x;            // b*NH + o
  const int b_ = bo / NH, o = bo % NH;
  const int t = threadIdx.x, lane = t & 31, wave = t >> 5;
  float weff[NH];
#pragma unroll
  for (int h = 0; h < NH; ++h)          // cat duplicates attn_g -> fold weights
    weff[h] = conv_w[o * 2 * NH + h] + conv_w[o * 2 * NH + NH + h];
  const float cb = conv_b[o];
  const _Float16* base = ag + (size_t)b_ * NH * SEQ * SEQ;
  _Float16* obase = at + (size_t)bo * SEQ * SEQ;
  for (int n = 0; n < SEQ; ++n) {
    float mval = cb;
#pragma unroll
    for (int h = 0; h < NH; ++h)
      mval += weff[h] * (float)base[(size_t)h * SEQ * SEQ + (size_t)n * SEQ + t];
    float mx = mval;
#pragma unroll
    for (int off = 16; off; off >>= 1) mx = fmaxf(mx, __shfl_xor(mx, off, 32));
    if (lane == 0) red[wave] = mx;
    __syncthreads();
    mx = red[0];
#pragma unroll
    for (int w = 1; w < 8; ++w) mx = fmaxf(mx, red[w]);
    __syncthreads();
    const float e = __expf(mval - mx);
    float sm = e;
#pragma unroll
    for (int off = 16; off; off >>= 1) sm += __shfl_xor(sm, off, 32);
    if (lane == 0) red[wave] = sm;
    __syncthreads();
    sm = 0.f;
#pragma unroll
    for (int w = 0; w < 8; ++w) sm += red[w];
    __syncthreads();
    obase[(size_t)n * SEQ + t] = (_Float16)(e / sm);
  }
}

// ---------------- K5: ctx = attn @ V; V slab staged into LDS by the TDM -----
// The V operand is K-major (B[k][n] = V[k*64+n]) -> 16-bit strided gathers if
// read from global. Instead one tensor_load_to_lds DMA (TENSORcnt-tracked)
// pulls the whole 256x64 f16 tile (32 KB) into LDS once per block; the 8 waves
// then feed their B fragments from LDS.
__global__ void __launch_bounds__(256)
k_attnv(const _Float16* __restrict__ at, const _Float16* __restrict__ vh,
        _Float16* __restrict__ ctx) {
  __shared__ _Float16 Vs[SEQ * HD];   // 32 KB, at LDS offset 0
  const int lane = threadIdx.x & 31, wave = threadIdx.x >> 5;
  const int slab = blockIdx.y;
  const int m0 = blockIdx.x * 32 + (wave & 1) * 16;
  const int n0 = (wave >> 1) * 16;      // 4 tiles cover HD=64
  const _Float16* A = at + (size_t)slab * SEQ * SEQ;
  const _Float16* V = vh + (size_t)slab * SEQ * HD;

  if (wave == 0) {                      // one TDM issue per block (EXEC ignored)
    const unsigned long long ga = (unsigned long long)(uintptr_t)V;
    u32x4 g0;
    i32x8 g1;
    i32x4 g2 = {0, 0, 0, 0}, g3 = {0, 0, 0, 0};
    i32x8 g4 = {0, 0, 0, 0, 0, 0, 0, 0};
    // D# group 0: count=1 | lds_addr | global_addr[56:0] | type=2 ("image")
    g0[0] = 1u;
    g0[1] = 0u;                         // Vs sits at LDS offset 0
    g0[2] = (unsigned)ga;
    g0[3] = ((unsigned)(ga >> 32) & 0x01FFFFFFu) | (2u << 30);
    // D# group 1: data_size=2B, tensor 64x256, tile 64x256, dim0_stride=64
    g1[0] = (int)(1u << 16);            // data_size = 1 -> 2 bytes
    g1[1] = (int)(64u << 16);           // tensor_dim0 = 64   (bits 79:48)
    g1[2] = (int)(256u << 16);          // tensor_dim1 = 256  (bits 111:80)
    g1[3] = (int)(64u << 16);           // tile_dim0 = 64     (bits 127:112)
    g1[4] = (int)256;                   // tile_dim1 = 256    (bits 143:128)
    g1[5] = (int)64;                    // tensor_dim0_stride (bits 207:160)
    g1[6] = 0;
    g1[7] = 0;
    // 6-arg (clang-23 / therock) form: extra int32x8 group before cpol
    __builtin_amdgcn_tensor_load_to_lds(g0, g1, g2, g3, g4, 0);
    __builtin_amdgcn_s_wait_tensorcnt((short)0);
  }
  __syncthreads();

  const int n = lane & 15, half = lane >> 4;
  v8f acc = {};
#pragma unroll 4
  for (int k0 = 0; k0 < SEQ; k0 += 32) {
    v16h a = ldA_f16(A, SEQ, m0, k0, lane);
    v16h b;
#pragma unroll
    for (int j = 0; j < 16; ++j)
      b[j] = Vs[(k0 + 16 * half + j) * HD + n0 + n];
    acc = __builtin_amdgcn_wmma_f32_16x16x32_f16(false, a, false, b, (short)0, acc, false, false);
  }
  const int dd = n0 + n;
  const int b_ = slab / NH, h = slab % NH;
#pragma unroll
  for (int v = 0; v < 8; ++v) {
    const int r = m0 + v + 8 * half;
    ctx[((size_t)b_ * SEQ + r) * DIMC + h * HD + dd] = (_Float16)acc[v];
  }
}

// ---------------- K6: out = ctx @ proj_w^T + proj_b (f32 result) ------------
__global__ void __launch_bounds__(256)
k_proj(const _Float16* __restrict__ ctx, const float* __restrict__ proj_w,
       const float* __restrict__ proj_b, float* __restrict__ out) {
  const int lane = threadIdx.x & 31, wave = threadIdx.x >> 5;
  const int m0 = blockIdx.y * 32 + (wave & 1) * 16;
  const int n0 = blockIdx.x * 64 + (wave >> 1) * 16;
  v8f acc = {};
#pragma unroll 4
  for (int k0 = 0; k0 < DIMC; k0 += 32) {
    if (k0 + 128 < DIMC) {
      __builtin_prefetch(ctx + (size_t)(m0 + (lane & 15)) * DIMC + k0 + 128, 0, 1);
      __builtin_prefetch(proj_w + (size_t)(n0 + (lane & 15)) * DIMC + k0 + 128, 0, 1);
    }
    v16h a = ldA_f16(ctx, DIMC, m0, k0, lane);
    v16h b = ldBT_f32(proj_w, DIMC, n0, k0, lane);
    acc = __builtin_amdgcn_wmma_f32_16x16x32_f16(false, a, false, b, (short)0, acc, false, false);
  }
  const int half = lane >> 4, col = n0 + (lane & 15);
  const float bias = proj_b[col];
#pragma unroll
  for (int v = 0; v < 8; ++v)
    out[(size_t)(m0 + v + 8 * half) * DIMC + col] = acc[v] + bias;
}

// ---------------- launcher ---------------------------------------------------
extern "C" void kernel_launch(void* const* d_in, const int* in_sizes, int n_in,
                              void* d_out, int out_size, void* d_ws, size_t ws_size,
                              hipStream_t stream) {
  (void)in_sizes; (void)n_in; (void)out_size; (void)ws_size;
  const float* x      = (const float*)d_in[0];
  const float* qkv_w  = (const float*)d_in[1];
  const float* qkv_b  = (const float*)d_in[2];
  const float* gsc    = (const float*)d_in[3];
  const float* conv_w = (const float*)d_in[4];
  const float* conv_b = (const float*)d_in[5];
  const float* proj_w = (const float*)d_in[6];
  const float* proj_b = (const float*)d_in[7];
  float* out = (float*)d_out;
  char*  ws  = (char*)d_ws;

  const size_t SLAB_ELEMS = (size_t)SLABS * SEQ * HD;   // 6,291,456
  const size_t ATT_ELEMS  = (size_t)SLABS * SEQ * SEQ;  // 25,165,824

  // Workspace layout (~132 MiB), with lifetime-safe aliasing:
  float*    qf  = (float*)(ws + 0);                          // 25.2 MB
  float*    kf  = (float*)(ws + SLAB_ELEMS * 4);             // 25.2 MB
  _Float16* ag  = (_Float16*)(ws + 0);                       // 50.3 MB (aliases qf/kf, dead by K3)
  _Float16* at  = (_Float16*)(ws + ATT_ELEMS * 2);           // 50.3 MB
  _Float16* vh  = (_Float16*)(ws + ATT_ELEMS * 4);           // 12.6 MB
  _Float16* qg  = (_Float16*)(ws + ATT_ELEMS * 4 + SLAB_ELEMS * 2);
  _Float16* kg  = (_Float16*)(ws + ATT_ELEMS * 4 + SLAB_ELEMS * 4);
  _Float16* ctx = qg;                                        // aliases qg (dead after K3)

  k_qkv       <<<dim3(3 * DIMC / 64, NB * SEQ / 32), 256, 0, stream>>>(x, qkv_w, qkv_b, qf, kf, vh);
  k_cholqr    <<<dim3(2 * SLABS),                    256, 0, stream>>>(qf, kf, qg, kg);
  k_dots      <<<dim3(SEQ / 64, SEQ / 32, SLABS),    256, 0, stream>>>(qg, kg, gsc, ag);
  k_mixsoftmax<<<dim3(SLABS),                        256, 0, stream>>>(ag, conv_w, conv_b, at);
  k_attnv     <<<dim3(SEQ / 32, SLABS),              256, 0, stream>>>(at, vh, ctx);
  k_proj      <<<dim3(DIMC / 64, NB * SEQ / 32),     256, 0, stream>>>(ctx, proj_w, proj_b, out);
}
